// Graph_Module_44092134261105
// MI455X (gfx1250) — matrix operand order repaired and checked by reading, not compiled, not run on hardware
//
#include <hip/hip_runtime.h>
#include <hip/hip_bf16.h>

// ---------------------------------------------------------------------------
// GCN with learned adjacency on MI455X (gfx1250, wave32, WMMA).
//   B=32, K=1024, IND=HID=OUT=256.
// All GEMMs via v_wmma_f32_16x16x32_bf16 (f32->bf16 staged in LDS, f32 acc).
// Workgroup tile 128x128, 8 waves (2x4), wave tile 64x32 (4x2 16x16 accs).
// Double-buffered LDS: global loads for tile t+1 issue before the WMMAs of
// tile t, ds_stores land in the other buffer, single barrier per K-step.
// All address math 32-bit (every slice < 2^31 bytes).
// ---------------------------------------------------------------------------

typedef __bf16 bf16;
typedef bf16  v4bf  __attribute__((ext_vector_type(4)));
typedef bf16  v8bf  __attribute__((ext_vector_type(8)));
typedef bf16  v16bf __attribute__((ext_vector_type(16)));
typedef float v8f   __attribute__((ext_vector_type(8)));

#define TILE_M 128
#define TILE_N 128
#define TILE_K 32
#define LDS_PAD 8              // row stride 40 bf16 = 80B: 16B aligned, conflict-free
#define LDS_STRIDE (TILE_K + LDS_PAD)

#define EPI_NONE      0
#define EPI_BIAS      1
#define EPI_BIAS_RELU 2

__device__ __forceinline__ v16bf ld_frag(const bf16* p0, const bf16* p1) {
    v8bf lo = *(const v8bf*)p0;   // ds_load_b128
    v8bf hi = *(const v8bf*)p1;   // ds_load_b128
    v16bf r;
#pragma unroll
    for (int i = 0; i < 8; ++i) { r[i] = lo[i]; r[i + 8] = hi[i]; }
    return r;
}

__device__ __forceinline__ v4bf pack4(float4 f) {
    v4bf p; p[0] = (bf16)f.x; p[1] = (bf16)f.y; p[2] = (bf16)f.z; p[3] = (bf16)f.w;
    return p;
}

// C[M,N] = epi( A[M,K] * op(B) + bias ),  op(B)=B[K,N] (BT=false) or B[N,K]^T (BT=true)
// All dims are multiples of the tile sizes for this workload; per-batch slices
// fit in 32-bit element offsets.
template <int EPI, bool BT>
__global__ __launch_bounds__(256) void gemm_bf16_wmma(
    const float* __restrict__ A, unsigned sA,
    const float* __restrict__ B, unsigned sB,
    const float* __restrict__ bias,
    float* __restrict__ C, unsigned sC,
    int M, int N, int Kdim)
{
    __shared__ alignas(16) bf16 As[2][TILE_M][LDS_STRIDE];   // [buf][m][k]
    __shared__ alignas(16) bf16 Bt[2][TILE_N][LDS_STRIDE];   // [buf][n][k] (N-major)

    A += (size_t)blockIdx.z * sA;
    B += (size_t)blockIdx.z * sB;
    C += (size_t)blockIdx.z * sC;

    const unsigned tM  = blockIdx.y * TILE_M;
    const unsigned tN  = blockIdx.x * TILE_N;
    const unsigned tid = threadIdx.x;
    const unsigned lane = tid & 31u;
    const unsigned wave = tid >> 5;
    const unsigned wm = wave >> 2;        // 0..1 -> 64-row band
    const unsigned wn = wave & 3u;        // 0..3 -> 32-col band

    v8f acc[4][2];
#pragma unroll
    for (int i = 0; i < 4; ++i)
#pragma unroll
        for (int j = 0; j < 2; ++j) acc[i][j] = (v8f){};

    float4 ra[4], rb[4];                  // staging registers

    auto load_tile = [&](unsigned k0) {
#pragma unroll
        for (int j = 0; j < 4; ++j) {
            unsigned idx = tid + j * 256u;
            unsigned row = idx >> 3;                       // 0..127
            unsigned c4  = (idx & 7u) << 2;                // 0..28
            ra[j] = *(const float4*)(A + (tM + row) * (unsigned)Kdim + k0 + c4);
        }
        if (BT) {
#pragma unroll
            for (int j = 0; j < 4; ++j) {
                unsigned idx = tid + j * 256u;
                unsigned row = idx >> 3;                   // n
                unsigned c4  = (idx & 7u) << 2;            // k
                rb[j] = *(const float4*)(B + (tN + row) * (unsigned)Kdim + k0 + c4);
            }
        } else {
#pragma unroll
            for (int j = 0; j < 4; ++j) {
                unsigned idx = tid + j * 256u;
                unsigned k  = idx >> 5;                    // 0..31
                unsigned n4 = (idx & 31u) << 2;            // 0..124
                rb[j] = *(const float4*)(B + (k0 + k) * (unsigned)N + tN + n4);
            }
        }
    };

    auto store_tile = [&](int buf) {
#pragma unroll
        for (int j = 0; j < 4; ++j) {
            unsigned idx = tid + j * 256u;
            unsigned row = idx >> 3;
            unsigned c4  = (idx & 7u) << 2;
            *(v4bf*)&As[buf][row][c4] = pack4(ra[j]);      // ds_store_b64
        }
        if (BT) {
#pragma unroll
            for (int j = 0; j < 4; ++j) {
                unsigned idx = tid + j * 256u;
                unsigned row = idx >> 3;
                unsigned c4  = (idx & 7u) << 2;
                *(v4bf*)&Bt[buf][row][c4] = pack4(rb[j]);  // ds_store_b64
            }
        } else {
#pragma unroll
            for (int j = 0; j < 4; ++j) {
                unsigned idx = tid + j * 256u;
                unsigned k  = idx >> 5;
                unsigned n4 = (idx & 31u) << 2;
                Bt[buf][n4 + 0][k] = (bf16)rb[j].x;        // transpose while staging
                Bt[buf][n4 + 1][k] = (bf16)rb[j].y;
                Bt[buf][n4 + 2][k] = (bf16)rb[j].z;
                Bt[buf][n4 + 3][k] = (bf16)rb[j].w;
            }
        }
    };

    auto compute = [&](int buf) {
        // B operand (32x16 bf16): lane n = lane&15, K = (lane>>4)*16 + 2v+h
        v16bf bfrag[2];
#pragma unroll
        for (int nt = 0; nt < 2; ++nt) {
            unsigned c  = wn * 32u + nt * 16u + (lane & 15u);
            unsigned kb = (lane >> 4) * 16u;
            bfrag[nt] = ld_frag(&Bt[buf][c][kb], &Bt[buf][c][kb + 8]);
        }
        // A operand (16x32 bf16): lane m = lane&15, K = (lane>>4)*16 + {0..15}
#pragma unroll
        for (int mt = 0; mt < 4; ++mt) {
            unsigned r  = wm * 64u + mt * 16u + (lane & 15u);
            unsigned kb = (lane >> 4) * 16u;
            v16bf afrag = ld_frag(&As[buf][r][kb], &As[buf][r][kb + 8]);
#pragma unroll
            for (int nt = 0; nt < 2; ++nt) {
                acc[mt][nt] = __builtin_amdgcn_wmma_f32_16x16x32_bf16(
                    false, afrag, false, bfrag[nt], (short)0, acc[mt][nt],
                    false, false);
            }
        }
    };

    const int nk = Kdim / TILE_K;
    load_tile(0);
    store_tile(0);
    __syncthreads();
    for (int t = 0; t < nk; ++t) {
        const bool has_next = (t + 1 < nk);
        if (has_next) load_tile((unsigned)(t + 1) * TILE_K);   // global latency hides
        compute(t & 1);                                        // under the WMMAs
        if (has_next) store_tile((t + 1) & 1);
        __syncthreads();
    }

    // Epilogue. C/D layout: lanes 0-15 -> M=r, lanes 16-31 -> M=8+r; N=lane&15.
#pragma unroll
    for (int mt = 0; mt < 4; ++mt) {
#pragma unroll
        for (int nt = 0; nt < 2; ++nt) {
            unsigned col = tN + wn * 32u + nt * 16u + (lane & 15u);
            unsigned rb0 = tM + wm * 64u + mt * 16u + (lane >> 4) * 8u;
            float bv = (EPI >= EPI_BIAS) ? bias[col] : 0.0f;
            float* cp = C + rb0 * (unsigned)N + col;
#pragma unroll
            for (int r = 0; r < 8; ++r) {
                float v = acc[mt][nt][r];
                if (EPI >= EPI_BIAS)      v += bv;
                if (EPI == EPI_BIAS_RELU) v  = fmaxf(v, 0.0f);
                *cp = v;
                cp += N;
            }
        }
    }
}

// One wave per row: d = rowsum(A)+1 (identity); dinv = d^-1/2.  K = 1024 fixed.
__global__ __launch_bounds__(256) void rowsum_dinv_kernel(
    const float* __restrict__ adj, float* __restrict__ dinv)
{
    unsigned rowid = (blockIdx.x * 256u + threadIdx.x) >> 5;   // wave id = row id
    unsigned lane  = threadIdx.x & 31u;
    const float* row = adj + (size_t)rowid * 1024u;
    float s = 0.0f;
#pragma unroll 4
    for (unsigned l = lane; l < 1024u; l += 32u) s += row[l];
#pragma unroll
    for (int off = 16; off; off >>= 1) s += __shfl_xor(s, off, 32);
    s += 1.0f;                       // +I on the diagonal
    if (lane == 0) dinv[rowid] = rsqrtf(s);
}

// adj[b,k,l] = dinv[b,k] * (A[b,k,l] + (k==l)) * dinv[b,l]   (in place, K=1024)
__global__ __launch_bounds__(256) void scale_adj_kernel(
    float* __restrict__ adj, const float* __restrict__ dinv)
{
    unsigned i = blockIdx.x * 256u + threadIdx.x;   // < 2^25, fits 32-bit
    unsigned rowid = i >> 10;                       // b*K + k
    unsigned l = i & 1023u;
    unsigned b = rowid >> 10;
    unsigned k = rowid & 1023u;
    float v = adj[i] + (k == l ? 1.0f : 0.0f);
    adj[i] = v * dinv[rowid] * dinv[(b << 10) + l];
}

extern "C" void kernel_launch(void* const* d_in, const int* in_sizes, int n_in,
                              void* d_out, int out_size, void* d_ws, size_t ws_size,
                              hipStream_t stream) {
    (void)in_sizes; (void)n_in; (void)out_size; (void)ws_size;

    constexpr int Bn = 32, Kn = 1024, D = 256;          // B, K, IND=HID=OUT
    constexpr unsigned BK   = (unsigned)Bn * Kn;        // 32768
    constexpr unsigned FEAT = BK * D;                   // 8,388,608 elems
    constexpr unsigned ADJ  = (unsigned)Bn * Kn * Kn;   // 33,554,432 elems

    const float* x    = (const float*)d_in[0];
    // d_in[1] = graph (empty -> learned adjacency path)
    const float* W1   = (const float*)d_in[2];
    const float* b1   = (const float*)d_in[3];
    const float* W2   = (const float*)d_in[4];
    const float* b2   = (const float*)d_in[5];
    const float* gc1W = (const float*)d_in[6];
    const float* gc1b = (const float*)d_in[7];
    const float* gc2W = (const float*)d_in[8];
    const float* gc2b = (const float*)d_in[9];

    float* out_adj   = (float*)d_out;                   // (B,K,K)
    float* out_feats = out_adj + ADJ;                   // (B,K,D)

    float* bufA = (float*)d_ws;                         // 32 MB
    float* bufB = bufA + FEAT;                          // 32 MB
    float* dinv = bufB + FEAT;                          // 128 KB

    const dim3 blk(256);
    const dim3 grid_flat(D / TILE_N, BK / TILE_M, 1);   // (2,256,1)
    const dim3 grid_kk(Kn / TILE_N, Kn / TILE_M, Bn);   // (8,8,32)
    const dim3 grid_kd(D / TILE_N, Kn / TILE_M, Bn);    // (2,8,32)

    // 1) h1 = relu(x @ W1 + b1)
    gemm_bf16_wmma<EPI_BIAS_RELU, false><<<grid_flat, blk, 0, stream>>>(
        x, 0, W1, 0, b1, bufA, 0, (int)BK, D, D);
    // 2) h2 = relu(h1 @ W2 + b2)
    gemm_bf16_wmma<EPI_BIAS_RELU, false><<<grid_flat, blk, 0, stream>>>(
        bufA, 0, W2, 0, b2, bufB, 0, (int)BK, D, D);
    // 3) Araw[b] = h2[b] @ h2[b]^T
    gemm_bf16_wmma<EPI_NONE, true><<<grid_kk, blk, 0, stream>>>(
        bufB, (unsigned)(Kn * D), bufB, (unsigned)(Kn * D), nullptr,
        out_adj, (unsigned)(Kn * Kn), Kn, Kn, D);
    // 4) dinv = (rowsum(Araw)+1)^-1/2
    rowsum_dinv_kernel<<<dim3(BK / 8), blk, 0, stream>>>(out_adj, dinv);
    // 5) adj = D^-1/2 (A+I) D^-1/2  (in place)
    scale_adj_kernel<<<dim3(ADJ / 256), blk, 0, stream>>>(out_adj, dinv);
    // 6) t1 = x @ gc1_W
    gemm_bf16_wmma<EPI_NONE, false><<<grid_flat, blk, 0, stream>>>(
        x, 0, gc1W, 0, nullptr, bufA, 0, (int)BK, D, D);
    // 7) hg = relu(adj @ t1 + gc1_b)
    gemm_bf16_wmma<EPI_BIAS_RELU, false><<<grid_kd, blk, 0, stream>>>(
        out_adj, (unsigned)(Kn * Kn), bufA, (unsigned)(Kn * D), gc1b,
        bufB, (unsigned)(Kn * D), Kn, D, Kn);
    // 8) t2 = hg @ gc2_W
    gemm_bf16_wmma<EPI_NONE, false><<<grid_flat, blk, 0, stream>>>(
        bufB, 0, gc2W, 0, nullptr, bufA, 0, (int)BK, D, D);
    // 9) feats = adj @ t2 + gc2_b
    gemm_bf16_wmma<EPI_BIAS, false><<<grid_kd, blk, 0, stream>>>(
        out_adj, (unsigned)(Kn * Kn), bufA, (unsigned)(Kn * D), gc2b,
        out_feats, (unsigned)(Kn * D), Kn, D, Kn);
}
